// PhaseBindingMemory_84825604096593
// MI455X (gfx1250) — matrix operand order repaired and checked by reading, compiled, not run on hardware
//
#include <hip/hip_runtime.h>
#include <math.h>
#include <stdint.h>

// ---------------------------------------------------------------------------
// Types for CDNA5 WMMA (wave32, 16x16x32 f16 -> f32 accum)
// ---------------------------------------------------------------------------
typedef _Float16 half8 __attribute__((ext_vector_type(8)));
typedef _Float16 v16h  __attribute__((ext_vector_type(16)));
typedef float    v8f   __attribute__((ext_vector_type(8)));

union FragU { v16h v; half8 h[2]; };

#define PI_F 3.14159265358979323846f

// A fragment: lane = row; half 0 -> K[0..15]; half 1 -> K[16..31]
__device__ inline void loadFragA(FragU& f, const _Float16* rowPtr, int k0, int hi) {
    const _Float16* p = rowPtr + k0 + hi * 16;
    f.h[0] = *(const half8*)(p);
    f.h[1] = *(const half8*)(p + 8);
}
// B fragment: lane = column; half 0 -> K[0..15]; half 1 -> K[16..31]
__device__ inline void loadFragB(FragU& f, const _Float16* colPtr, int k0, int hi) {
    const _Float16* p = colPtr + k0 + hi * 16;
    f.h[0] = *(const half8*)(p);
    f.h[1] = *(const half8*)(p + 8);
}

__device__ inline v8f wmma16(const FragU& a, const FragU& b, v8f c) {
    return __builtin_amdgcn_wmma_f32_16x16x32_f16(false, a.v, false, b.v,
                                                  (short)0, c, false, false);
}

// ---------------------------------------------------------------------------
// f32 -> f16 convert
// ---------------------------------------------------------------------------
__global__ void cvt_f32_f16(const float* __restrict__ src, _Float16* __restrict__ dst, int n) {
    int i = blockIdx.x * blockDim.x + threadIdx.x;
    if (i < n) dst[i] = (_Float16)src[i];
}

// f32 [R,C] -> f16 transposed [C,R]
__global__ void cvt_transpose_f16(const float* __restrict__ src, _Float16* __restrict__ dst,
                                  int R, int C) {
    int i = blockIdx.x * blockDim.x + threadIdx.x;
    if (i >= R * C) return;
    int r = i / C, c = i % C;
    dst[c * R + r] = (_Float16)src[i];
}

// ---------------------------------------------------------------------------
// LDS-staged WMMA GEMM for Nd-wide outputs, Kd = 512 fixed.
//   out = act(A[M,512] * BT[Nd,512]^T + bias[Nd])
// Block: 256 threads (8 waves). Block tile: 256 rows x 64 cols.
// Weight panel (64 cols x 512 k, f16) loaded once into LDS with
// GLOBAL_LOAD_ASYNC_TO_LDS_B128 (ASYNCcnt), then reused by all 8 waves.
// Each wave: 32 rows x 64 cols = 8 accumulators; A is register double-buffered.
// Grid: (M/256, Nd/64).
// ---------------------------------------------------------------------------
template <int ACT, bool OUTF32>
__global__ __launch_bounds__(256)
void gemm_f16_lds(const _Float16* __restrict__ A,
                  const _Float16* __restrict__ BT,
                  const float* __restrict__ bias,
                  _Float16* __restrict__ outH,
                  float* __restrict__ outF,
                  int Nd) {
    constexpr int KD  = 512;
    constexpr int LDB = KD + 8;                       // pad 8 halves: kills bank-stride conflicts
    __shared__ __align__(16) _Float16 sB[64 * LDB];   // ~65 KB

    const int tid    = threadIdx.x;
    const int wave   = tid >> 5;
    const int lane   = tid & 31;
    const int laneLo = lane & 15;
    const int hi     = lane >> 4;

    const int n0 = blockIdx.y * 64;
    const int m0 = blockIdx.x * 256 + wave * 32;

    // ---- async fill of the 64x512 B panel into LDS (4096 x 16B chunks) ----
    {
        const uint32_t ldsBase = (uint32_t)(uintptr_t)&sB[0];
        const _Float16* gbase  = BT + (size_t)n0 * KD;
#pragma unroll
        for (int i = 0; i < 16; ++i) {
            int id = i * 256 + tid;                 // chunk id
            int c  = id >> 6;                       // column 0..63
            int j  = id & 63;                       // 16B chunk within column
            const _Float16* ga = gbase + (size_t)c * KD + j * 8;
            uint32_t la = ldsBase + (uint32_t)(c * (LDB * 2) + j * 16);
            asm volatile("global_load_async_to_lds_b128 %0, %1, off"
                         :: "v"(la), "v"(ga) : "memory");
        }
        asm volatile("s_wait_asynccnt 0x0" ::: "memory");
    }
    __syncthreads();

    const _Float16* aRow0 = A + (size_t)(m0 + laneLo) * KD;
    const _Float16* aRow1 = aRow0 + (size_t)16 * KD;

    v8f acc[2][4];
#pragma unroll
    for (int mt = 0; mt < 2; ++mt)
#pragma unroll
        for (int t = 0; t < 4; ++t)
            acc[mt][t] = (v8f){0.f, 0.f, 0.f, 0.f, 0.f, 0.f, 0.f, 0.f};

    FragU a0c, a1c, a0n, a1n;
    loadFragA(a0c, aRow0, 0, hi);
    loadFragA(a1c, aRow1, 0, hi);

    for (int k0 = 0; k0 < KD; k0 += 32) {
        if (k0 + 32 < KD) {                         // prefetch next A tiles (reg double buffer)
            loadFragA(a0n, aRow0, k0 + 32, hi);
            loadFragA(a1n, aRow1, k0 + 32, hi);
        }
#pragma unroll
        for (int t = 0; t < 4; ++t) {
            FragU b;
            const _Float16* pb = &sB[(t * 16 + laneLo) * LDB + k0 + hi * 16];
            b.h[0] = *(const half8*)(pb);
            b.h[1] = *(const half8*)(pb + 8);
            acc[0][t] = wmma16(a0c, b, acc[0][t]);
            acc[1][t] = wmma16(a1c, b, acc[1][t]);
        }
        a0c = a0n;
        a1c = a1n;
    }

    // ---- epilogue: bias + act, f16 or f32 out ----
#pragma unroll
    for (int mt = 0; mt < 2; ++mt) {
        const int rowBase = m0 + mt * 16 + hi * 8;
#pragma unroll
        for (int t = 0; t < 4; ++t) {
            const int col = n0 + t * 16 + laneLo;
            const float bv = bias[col];
#pragma unroll
            for (int r = 0; r < 8; ++r) {
                float x = acc[mt][t][r] + bv;
                if (ACT == 1) x = tanhf(x);
                const size_t o = (size_t)(rowBase + r) * Nd + col;
                if (OUTF32) outF[o] = x;
                else        outH[o] = (_Float16)x;
            }
        }
    }
}

// ---------------------------------------------------------------------------
// Pipelined direct-global GEMM for Nd = 32 (phase / amplitude projections).
// One wave computes 16 rows x 32 cols; A and B fragments register
// double-buffered so WMMAs overlap the next iteration's loads.
// ---------------------------------------------------------------------------
template <int ACT>
__global__ void gemm_f16_small(const _Float16* __restrict__ A,
                               const _Float16* __restrict__ BT,
                               const float* __restrict__ bias,
                               _Float16* __restrict__ outH,
                               int M, int Kd) {
    constexpr int Nd = 32;
    const int wave = (blockIdx.x * blockDim.x + threadIdx.x) >> 5;
    const int lane = threadIdx.x & 31;
    if (wave >= M / 16) return;                      // whole-wave exit, EXEC stays all-ones

    const int laneLo = lane & 15;
    const int hi = lane >> 4;
    const int m0 = wave * 16;

    const _Float16* aRow  = A + (size_t)(m0 + laneLo) * Kd;
    const _Float16* bCol0 = BT + (size_t)laneLo * Kd;
    const _Float16* bCol1 = BT + (size_t)(16 + laneLo) * Kd;

    v8f acc0 = (v8f){0.f, 0.f, 0.f, 0.f, 0.f, 0.f, 0.f, 0.f};
    v8f acc1 = acc0;

    FragU ac, b0c, b1c, an, b0n, b1n;
    loadFragA(ac, aRow, 0, hi);
    loadFragB(b0c, bCol0, 0, hi);
    loadFragB(b1c, bCol1, 0, hi);

    for (int k0 = 32; k0 < Kd; k0 += 32) {
        loadFragA(an, aRow, k0, hi);
        loadFragB(b0n, bCol0, k0, hi);
        loadFragB(b1n, bCol1, k0, hi);
        acc0 = wmma16(ac, b0c, acc0);
        acc1 = wmma16(ac, b1c, acc1);
        ac = an; b0c = b0n; b1c = b1n;
    }
    acc0 = wmma16(ac, b0c, acc0);
    acc1 = wmma16(ac, b1c, acc1);

    const int rowBase = m0 + hi * 8;
    {
        const float bv0 = bias[laneLo];
        const float bv1 = bias[16 + laneLo];
#pragma unroll
        for (int r = 0; r < 8; ++r) {
            float x0 = acc0[r] + bv0;
            float x1 = acc1[r] + bv1;
            if (ACT == 1) { x0 = tanhf(x0); x1 = tanhf(x1); }
            outH[(size_t)(rowBase + r) * Nd + laneLo]      = (_Float16)x0;
            outH[(size_t)(rowBase + r) * Nd + 16 + laneLo] = (_Float16)x1;
        }
    }
}

// ---------------------------------------------------------------------------
// Phasor: re = (softplus(a)+0.1)*cos(tanh(p)*pi), im = ...*sin(...)
// transposed=1 -> out [B,32,perB] (keys); transposed=0 -> out [rows,32]
// ---------------------------------------------------------------------------
__global__ void phasor_kernel(const _Float16* __restrict__ Plin,
                              const _Float16* __restrict__ Alin,
                              float* __restrict__ outRe, float* __restrict__ outIm,
                              int rows, int perB, int transposed) {
    int i = blockIdx.x * blockDim.x + threadIdx.x;
    if (i >= rows * 32) return;
    int row = i >> 5, k = i & 31;
    float p = tanhf((float)Plin[i]) * PI_F;
    float a = (float)Alin[i];
    float sp = (a > 20.f) ? a : log1pf(expf(a));
    float amp = sp + 0.1f;
    float re = amp * cosf(p);
    float im = amp * sinf(p);
    size_t o;
    if (transposed) {
        int b = row / perB, n = row % perB;
        o = ((size_t)(b * 32 + k)) * perB + n;
    } else {
        o = (size_t)i;
    }
    outRe[o] = re;
    outIm[o] = im;
}

// ---------------------------------------------------------------------------
// mem partials: preP[b,s,k,d] = sum_{n in chunk s} kreT[b,k,n] * V[b,n,d]
// grid = (2, 32, B*NS), block = 256
// ---------------------------------------------------------------------------
__global__ void mem_partial_kernel(const float* __restrict__ kreT,
                                   const float* __restrict__ kimT,
                                   const _Float16* __restrict__ V,
                                   float* __restrict__ preP, float* __restrict__ pimP,
                                   int N, int NS) {
    const int d = blockIdx.x * 256 + threadIdx.x;
    const int kidx = blockIdx.y;
    const int b = blockIdx.z / NS;
    const int s = blockIdx.z % NS;
    const int chunk = N / NS;
    const int n0 = s * chunk;

    const float* wr = kreT + ((size_t)(b * 32 + kidx)) * N;
    const float* wi = kimT + ((size_t)(b * 32 + kidx)) * N;
    const _Float16* vb = V + ((size_t)b * N) * 512;

    float accRe = 0.f, accIm = 0.f;
    for (int n = n0; n < n0 + chunk; ++n) {
        float v = (float)vb[(size_t)n * 512 + d];
        accRe += wr[n] * v;
        accIm += wi[n] * v;
    }
    const size_t o = ((size_t)((b * NS + s) * 32 + kidx)) * 512 + d;
    preP[o] = accRe;
    pimP[o] = accIm;
}

// Deterministic reduce over NS splits -> mem[b,k,d]
__global__ void mem_reduce_kernel(const float* __restrict__ preP,
                                  const float* __restrict__ pimP,
                                  float* __restrict__ memRe, float* __restrict__ memIm,
                                  int NS) {
    int i = blockIdx.x * blockDim.x + threadIdx.x;   // (b,k,d): 4*32*512
    if (i >= 4 * 32 * 512) return;
    int b = i / (32 * 512);
    int kd = i % (32 * 512);
    float sr = 0.f, si = 0.f;
    for (int s = 0; s < NS; ++s) {
        size_t o = ((size_t)(b * NS + s)) * 32 * 512 + kd;
        sr += preP[o];
        si += pimP[o];
    }
    memRe[i] = sr;
    memIm[i] = si;
}

// ---------------------------------------------------------------------------
// Fused retrieve (K=32) + scale + LayerNorm; writes xhat*g+b as f16.
// One block (256 threads) per output row; each thread owns 2 columns.
// ---------------------------------------------------------------------------
__global__ void retrieve_ln_kernel(const float* __restrict__ qre,
                                   const float* __restrict__ qim,
                                   const float* __restrict__ memRe,
                                   const float* __restrict__ memIm,
                                   const float* __restrict__ ln_g,
                                   const float* __restrict__ ln_b,
                                   _Float16* __restrict__ xhat,
                                   float scale, int Q) {
    const int row = blockIdx.x;              // b*Q + q
    const int b = row / Q;
    const int t = threadIdx.x;

    const float* qr = qre + (size_t)row * 32;
    const float* qi = qim + (size_t)row * 32;
    const float* mr = memRe + (size_t)b * 32 * 512;
    const float* mi = memIm + (size_t)b * 32 * 512;

    float a0 = 0.f, a1 = 0.f;
#pragma unroll
    for (int k = 0; k < 32; ++k) {
        float cr = qr[k], ci = qi[k];
        a0 += cr * mr[k * 512 + t]       + ci * mi[k * 512 + t];
        a1 += cr * mr[k * 512 + t + 256] + ci * mi[k * 512 + t + 256];
    }
    a0 *= scale; a1 *= scale;

    __shared__ float sSum[256], sSq[256];
    sSum[t] = a0 + a1;
    sSq[t]  = a0 * a0 + a1 * a1;
    __syncthreads();
    for (int s = 128; s > 0; s >>= 1) {
        if (t < s) { sSum[t] += sSum[t + s]; sSq[t] += sSq[t + s]; }
        __syncthreads();
    }
    const float mean = sSum[0] * (1.f / 512.f);
    const float var  = sSq[0] * (1.f / 512.f) - mean * mean;
    const float rstd = rsqrtf(var + 1e-5f);

    xhat[(size_t)row * 512 + t]       = (_Float16)((a0 - mean) * rstd * ln_g[t] + ln_b[t]);
    xhat[(size_t)row * 512 + t + 256] = (_Float16)((a1 - mean) * rstd * ln_g[t + 256] + ln_b[t + 256]);
}

// ---------------------------------------------------------------------------
// Host orchestration
// ---------------------------------------------------------------------------
extern "C" void kernel_launch(void* const* d_in, const int* in_sizes, int n_in,
                              void* d_out, int out_size, void* d_ws, size_t ws_size,
                              hipStream_t stream) {
    (void)in_sizes; (void)n_in; (void)out_size; (void)ws_size;

    constexpr int B = 4, N = 8192, Q = 2048, D = 512, KP = 32;
    constexpr int BN = B * N;    // 32768
    constexpr int BQ = B * Q;    // 8192
    constexpr int NS = 16;       // split-N factor for mem reduction

    const float* keys    = (const float*)d_in[0];
    const float* values  = (const float*)d_in[1];
    const float* queries = (const float*)d_in[2];
    const float* Wp1_k   = (const float*)d_in[3];
    const float* bp1_k   = (const float*)d_in[4];
    const float* Wp2_k   = (const float*)d_in[5];
    const float* bp2_k   = (const float*)d_in[6];
    const float* Wa_k    = (const float*)d_in[7];
    const float* ba_k    = (const float*)d_in[8];
    const float* Wp1_q   = (const float*)d_in[9];
    const float* bp1_q   = (const float*)d_in[10];
    const float* Wp2_q   = (const float*)d_in[11];
    const float* bp2_q   = (const float*)d_in[12];
    const float* Wa_q    = (const float*)d_in[13];
    const float* ba_q    = (const float*)d_in[14];
    const float* Wv      = (const float*)d_in[15];
    const float* bv      = (const float*)d_in[16];
    const float* ln_g    = (const float*)d_in[17];
    const float* ln_b    = (const float*)d_in[18];
    const float* Wo      = (const float*)d_in[19];
    const float* bo      = (const float*)d_in[20];
    float* out = (float*)d_out;

    // ---- workspace carve-out (aligned) ----
    char* ws = (char*)d_ws;
    size_t off = 0;
    auto carve = [&](size_t bytes) -> void* {
        off = (off + 255) & ~(size_t)255;
        void* p = ws + off;
        off += bytes;
        return p;
    };
    _Float16* bufA   = (_Float16*)carve((size_t)BN * D * 2);   // keysH / valuesH / queriesH / xhat
    _Float16* bufB   = (_Float16*)carve((size_t)BN * D * 2);   // hidden / V / hiddenQ
    _Float16* Plin   = (_Float16*)carve((size_t)BN * KP * 2);
    _Float16* Alin   = (_Float16*)carve((size_t)BN * KP * 2);
    float*    kreT   = (float*)carve((size_t)B * KP * N * 4);  // [B,32,N]
    float*    kimT   = (float*)carve((size_t)B * KP * N * 4);
    float*    qre    = (float*)carve((size_t)BQ * KP * 4);     // [B,Q,32]
    float*    qim    = (float*)carve((size_t)BQ * KP * 4);
    float*    preP   = (float*)carve((size_t)B * NS * KP * D * 4);
    float*    pimP   = (float*)carve((size_t)B * NS * KP * D * 4);
    float*    memRe  = (float*)carve((size_t)B * KP * D * 4);
    float*    memIm  = (float*)carve((size_t)B * KP * D * 4);
    _Float16* Wp1kT  = (_Float16*)carve((size_t)D * D * 2);
    _Float16* Wp2kT  = (_Float16*)carve((size_t)D * KP * 2);
    _Float16* WakT   = (_Float16*)carve((size_t)D * KP * 2);
    _Float16* Wp1qT  = (_Float16*)carve((size_t)D * D * 2);
    _Float16* Wp2qT  = (_Float16*)carve((size_t)D * KP * 2);
    _Float16* WaqT   = (_Float16*)carve((size_t)D * KP * 2);
    _Float16* WvT    = (_Float16*)carve((size_t)D * D * 2);
    _Float16* WoT    = (_Float16*)carve((size_t)D * D * 2);

    auto cvtN = [&](const float* s, _Float16* d, int n) {
        cvt_f32_f16<<<(n + 255) / 256, 256, 0, stream>>>(s, d, n);
    };
    auto cvtT = [&](const float* s, _Float16* d, int R, int C) {
        cvt_transpose_f16<<<(R * C + 255) / 256, 256, 0, stream>>>(s, d, R, C);
    };
    auto smallBlocks = [](int M) { return ((M / 16) + 7) / 8; };

    // ---- weights: convert + transpose ----
    cvtT(Wp1_k, Wp1kT, D, D);
    cvtT(Wp2_k, Wp2kT, D, KP);
    cvtT(Wa_k,  WakT,  D, KP);
    cvtT(Wp1_q, Wp1qT, D, D);
    cvtT(Wp2_q, Wp2qT, D, KP);
    cvtT(Wa_q,  WaqT,  D, KP);
    cvtT(Wv,    WvT,   D, D);
    cvtT(Wo,    WoT,   D, D);

    const dim3 gBN(BN / 256, D / 64);   // (128, 8)
    const dim3 gBQ(BQ / 256, D / 64);   // (32, 8)

    // ---- keys path ----
    cvtN(keys, bufA, BN * D);
    gemm_f16_lds<1, false><<<gBN, 256, 0, stream>>>(
        bufA, Wp1kT, bp1_k, bufB, nullptr, D);                        // hidden = tanh(x@Wp1+b)
    gemm_f16_small<0><<<smallBlocks(BN), 256, 0, stream>>>(
        bufB, Wp2kT, bp2_k, Plin, BN, D);                             // phase linear
    gemm_f16_small<0><<<smallBlocks(BN), 256, 0, stream>>>(
        bufA, WakT, ba_k, Alin, BN, D);                               // amp linear
    phasor_kernel<<<(BN * KP + 255) / 256, 256, 0, stream>>>(
        Plin, Alin, kreT, kimT, BN, N, 1);                            // -> [B,32,N]

    // ---- value projection ----
    cvtN(values, bufA, BN * D);
    gemm_f16_lds<0, false><<<gBN, 256, 0, stream>>>(
        bufA, WvT, bv, bufB, nullptr, D);                             // V (f16) in bufB

    // ---- memory binding: mem = kre^T @ V (split-N, deterministic reduce) ----
    {
        dim3 g(2, KP, B * NS);
        mem_partial_kernel<<<g, 256, 0, stream>>>(kreT, kimT, bufB, preP, pimP, N, NS);
        mem_reduce_kernel<<<(B * KP * D + 255) / 256, 256, 0, stream>>>(
            preP, pimP, memRe, memIm, NS);
    }

    // ---- queries path ----
    cvtN(queries, bufA, BQ * D);
    gemm_f16_lds<1, false><<<gBQ, 256, 0, stream>>>(
        bufA, Wp1qT, bp1_q, bufB, nullptr, D);
    gemm_f16_small<0><<<smallBlocks(BQ), 256, 0, stream>>>(
        bufB, Wp2qT, bp2_q, Plin, BQ, D);
    gemm_f16_small<0><<<smallBlocks(BQ), 256, 0, stream>>>(
        bufA, WaqT, ba_q, Alin, BQ, D);
    phasor_kernel<<<(BQ * KP + 255) / 256, 256, 0, stream>>>(
        Plin, Alin, qre, qim, BQ, Q, 0);                              // -> [B,Q,32]

    // ---- fused retrieve + scale + LayerNorm -> f16 xhat (in bufA) ----
    retrieve_ln_kernel<<<BQ, 256, 0, stream>>>(
        qre, qim, memRe, memIm, ln_g, ln_b, bufA, 1.0f / 512.0f, Q);  // 1/sqrt(N*K)=1/512

    // ---- output head: out = xhat @ Wo + bo (f32 out) ----
    gemm_f16_lds<0, true><<<gBQ, 256, 0, stream>>>(
        bufA, WoT, bo, nullptr, out, D);
}
